// Block_6476810682806
// MI455X (gfx1250) — matrix operand order repaired and checked by reading, not compiled, not run on hardware
//
#include <hip/hip_runtime.h>

// CDNA5 (gfx1250) fused kervolution bottleneck:
//   3 WMMA-f16 GEMM stages, (y+1)^2 * bn_scale + bn_shift fused epilogues,
//   f16 intermediates in workspace (L2-resident), residual fused in stage 3.
//   N-tile = 32 pixels (2 B-fragments per A-fragment -> 8 WMMA / 4 A-loads per
//   K-chunk per wave), branch-free clamped tail loads, pair-packed LDS stores.

typedef __attribute__((ext_vector_type(16))) _Float16 v16h;
typedef __attribute__((ext_vector_type(4)))  _Float16 v4h;
typedef __attribute__((ext_vector_type(2)))  _Float16 v2h;
typedef __attribute__((ext_vector_type(8)))  float    v8f;
typedef __attribute__((ext_vector_type(4)))  float    v4f;

#define HW   196          // 14*14 pixels per image per channel
#define NT32 7            // ceil(196/32) N-tiles of 32 pixels
#define LDSP 40           // LDS row stride in halves (32 + 8 pad, conflict-free)

__device__ __forceinline__ v8f wmma_f16(v16h a, v16h b, v8f c) {
  return __builtin_amdgcn_wmma_f32_16x16x32_f16(
      /*neg_a=*/false, a, /*neg_b=*/false, b,
      /*c_mod=*/(short)0, c, /*reuse_a=*/false, /*reuse_b=*/false);
}

// K offset inside a 32-wide chunk for (lane, half j): contiguous, K = 16*(lane/16) + j
__device__ __forceinline__ int kOff(int lane, int j) {
  return (lane < 16) ? j          // K 0..15
                     : j + 16;    // K 16..31
}

// ---- pre-pack weight [M,Ktot] (f32, row-major) into WMMA A-fragment order --
__global__ __launch_bounds__(256)
void pack_w(const float* __restrict__ W, _Float16* __restrict__ P,
            int KC, int Ktot, int total) {
  int idx = blockIdx.x * 256 + threadIdx.x;
  if (idx >= total) return;
  int tile = idx >> 9;
  int rem  = idx & 511;
  int lane = rem >> 4;
  int j    = rem & 15;
  int mt = tile / KC, kc = tile - mt * KC;
  int M = mt * 16 + (lane & 15);
  int K = kc * 32 + kOff(lane, j);
  P[idx] = (_Float16)W[(size_t)M * Ktot + K];
}

// ---- fold inference BatchNorm to per-channel scale/shift -------------------
__global__ __launch_bounds__(256)
void bn_fold(const float* __restrict__ g, const float* __restrict__ b,
             const float* __restrict__ m, const float* __restrict__ v,
             float* __restrict__ sc, float* __restrict__ sh, int C) {
  int c = blockIdx.x * 256 + threadIdx.x;
  if (c >= C) return;
  float inv = g[c] * rsqrtf(v[c] + 1e-5f);
  sc[c] = inv;
  sh[c] = b[c] - m[c] * inv;
}

// ============================================================================
// Stage 1: h1 = pointwise( W1[256,1024] @ X_n[1024,196] )   (f32 in, f16 out)
// ============================================================================
__global__ __launch_bounds__(128)
void kerv_gemm1(const float* __restrict__ X, const _Float16* __restrict__ pA,
                const float* __restrict__ sc, const float* __restrict__ sh,
                _Float16* __restrict__ H1) {
  constexpr int KC = 32;                       // 1024 / 32
  __shared__ _Float16 ldsB[32 * LDSP];
  const int nt = blockIdx.x, n = blockIdx.y;
  const int p0 = nt * 32;
  const int t = threadIdx.x;
  const int lane = t & 31;
  const int mtBase = (t >> 5) * 4;             // 4 waves x 4 M-tiles = M 0..255
  const float* Xn = X + (size_t)n * 1024 * HW;
  const int k0  = (t >> 3) * 2;                // even K row pair this thread stages
  const int nc0 = (t & 7) * 4;                 // pixel sub-column (0,4,...,28)
  v8f acc[4][2] = {};

  for (int kc = 0; kc < KC; ++kc) {
    // --- stage B tile [32K x 32N] into LDS transposed, pair-packed ---
    const float* row0 = Xn + (size_t)(kc * 32 + k0) * HW;
    const float* row1 = row0 + HW;
    float a0[4], a1[4];
    if (p0 + 32 <= HW) {                       // aligned float4 fast path
      v4f u0 = *(const v4f*)(row0 + p0 + nc0);
      v4f u1 = *(const v4f*)(row1 + p0 + nc0);
#pragma unroll
      for (int j = 0; j < 4; ++j) { a0[j] = u0[j]; a1[j] = u1[j]; }
    } else {                                   // branch-free clamped tail
#pragma unroll
      for (int j = 0; j < 4; ++j) {
        int p = p0 + nc0 + j;
        int pc = p < HW ? p : HW - 1;
        float x0 = row0[pc], x1 = row1[pc];
        a0[j] = p < HW ? x0 : 0.0f;
        a1[j] = p < HW ? x1 : 0.0f;
      }
    }
#pragma unroll
    for (int j = 0; j < 4; ++j) {
      v2h pr; pr[0] = (_Float16)a0[j]; pr[1] = (_Float16)a1[j];
      *(v2h*)&ldsB[(nc0 + j) * LDSP + k0] = pr;    // ds_store_b32
    }
    __syncthreads();

    // --- fragments + 8 WMMAs per wave (A reused across 2 N-halves) ---
    v16h bf0 = *(const v16h*)&ldsB[(lane & 15) * LDSP + ((lane >> 4) << 4)];
    v16h bf1 = *(const v16h*)&ldsB[(16 + (lane & 15)) * LDSP + ((lane >> 4) << 4)];
    const _Float16* aB = pA + ((size_t)mtBase * KC + kc) * 512 + lane * 16;
#pragma unroll
    for (int i = 0; i < 4; ++i) {
      v16h af = *(const v16h*)(aB + (size_t)i * KC * 512);
      acc[i][0] = wmma_f16(af, bf0, acc[i][0]);
      acc[i][1] = wmma_f16(af, bf1, acc[i][1]);
    }
    __syncthreads();
  }

  // --- fused epilogue: (y+1)^2 * sc + sh, store f16 ---
  const int ncl = lane & 15;
  const int mh  = (lane >> 4) * 8;
  const bool ok0 = p0 + ncl < HW, ok1 = p0 + 16 + ncl < HW;
#pragma unroll
  for (int i = 0; i < 4; ++i)
#pragma unroll
    for (int r = 0; r < 8; ++r) {
      int cc = (mtBase + i) * 16 + mh + r;
      float scv = sc[cc], shv = sh[cc];
      _Float16* dst = H1 + ((size_t)n * 256 + cc) * HW + p0 + ncl;
      if (ok0) { float y = acc[i][0][r] + 1.0f; dst[0]  = (_Float16)(y * y * scv + shv); }
      if (ok1) { float y = acc[i][1][r] + 1.0f; dst[16] = (_Float16)(y * y * scv + shv); }
    }
}

// ============================================================================
// Stage 2: h2 = pointwise( W2[256,2304] @ im2col(h1)[2304,196] )  (f16 in/out)
// ============================================================================
__global__ __launch_bounds__(128)
void kerv_gemm2(const _Float16* __restrict__ H1, const _Float16* __restrict__ pA,
                const float* __restrict__ sc, const float* __restrict__ sh,
                _Float16* __restrict__ H2) {
  constexpr int KC = 72;                       // 2304 / 32, K = (c,ky,kx)
  __shared__ _Float16 ldsB[32 * LDSP];
  __shared__ unsigned imtab[2304];             // per-K: base offset + (ky,kx)
  const int nt = blockIdx.x, n = blockIdx.y;
  const int p0 = nt * 32;
  const int t = threadIdx.x;
  const int lane = t & 31;
  const int mtBase = (t >> 5) * 4;
  const _Float16* H1n = H1 + (size_t)n * 256 * HW;
  const int k0  = (t >> 3) * 2;
  const int nc0 = (t & 7) * 4;
  v8f acc[4][2] = {};

  // build im2col table once: K -> c*HW + (ky-1)*14 + (kx-1) (+32 bias), ky, kx
  for (int i = t; i < 2304; i += 128) {
    int c = i / 9, r = i - 9 * c;
    int ky = r / 3, kx = r - 3 * ky;
    int off = c * HW + (ky - 1) * 14 + (kx - 1) + 32;   // >= 17, fits 18 bits
    imtab[i] = (unsigned)off | ((unsigned)ky << 18) | ((unsigned)kx << 20);
  }
  // per-thread pixel precompute (constant across K-chunks)
  int pcl[4], ph[4], pw[4]; bool pv[4];
#pragma unroll
  for (int j = 0; j < 4; ++j) {
    int p = p0 + nc0 + j;
    pv[j] = p < HW;
    int pc = pv[j] ? p : HW - 1;
    pcl[j] = pc; ph[j] = pc / 14; pw[j] = pc - 14 * ph[j];
  }
  __syncthreads();

  for (int kc = 0; kc < KC; ++kc) {
    // --- branch-free im2col gather -> transposed LDS tile ---
    _Float16 vals[2][4];
#pragma unroll
    for (int kk = 0; kk < 2; ++kk) {
      unsigned e = imtab[kc * 32 + k0 + kk];
      int off = (int)(e & 0x3FFFFu) - 32;
      int ky = (e >> 18) & 3, kx = (e >> 20) & 3;
#pragma unroll
      for (int j = 0; j < 4; ++j) {
        int idx = off + pcl[j];
        idx = idx < 0 ? 0 : idx;
        idx = idx > 256 * HW - 1 ? 256 * HW - 1 : idx;
        _Float16 v = H1n[idx];                 // unconditional clamped load
        bool ok = pv[j] && ((unsigned)(ph[j] + ky - 1) < 14u)
                        && ((unsigned)(pw[j] + kx - 1) < 14u);
        vals[kk][j] = ok ? v : (_Float16)0.0f;
      }
    }
#pragma unroll
    for (int j = 0; j < 4; ++j) {
      v2h pr; pr[0] = vals[0][j]; pr[1] = vals[1][j];
      *(v2h*)&ldsB[(nc0 + j) * LDSP + k0] = pr;
    }
    __syncthreads();

    v16h bf0 = *(const v16h*)&ldsB[(lane & 15) * LDSP + ((lane >> 4) << 4)];
    v16h bf1 = *(const v16h*)&ldsB[(16 + (lane & 15)) * LDSP + ((lane >> 4) << 4)];
    const _Float16* aB = pA + ((size_t)mtBase * KC + kc) * 512 + lane * 16;
#pragma unroll
    for (int i = 0; i < 4; ++i) {
      v16h af = *(const v16h*)(aB + (size_t)i * KC * 512);
      acc[i][0] = wmma_f16(af, bf0, acc[i][0]);
      acc[i][1] = wmma_f16(af, bf1, acc[i][1]);
    }
    __syncthreads();
  }

  const int ncl = lane & 15;
  const int mh  = (lane >> 4) * 8;
  const bool ok0 = p0 + ncl < HW, ok1 = p0 + 16 + ncl < HW;
#pragma unroll
  for (int i = 0; i < 4; ++i)
#pragma unroll
    for (int r = 0; r < 8; ++r) {
      int cc = (mtBase + i) * 16 + mh + r;
      float scv = sc[cc], shv = sh[cc];
      _Float16* dst = H2 + ((size_t)n * 256 + cc) * HW + p0 + ncl;
      if (ok0) { float y = acc[i][0][r] + 1.0f; dst[0]  = (_Float16)(y * y * scv + shv); }
      if (ok1) { float y = acc[i][1][r] + 1.0f; dst[16] = (_Float16)(y * y * scv + shv); }
    }
}

// ============================================================================
// Stage 3: out = pointwise( W3[1024,256] @ h2[256,196] ) + x   (f32 out)
// grid.z in [0,4) splits the 1024 output channels.
// ============================================================================
__global__ __launch_bounds__(128)
void kerv_gemm3(const _Float16* __restrict__ H2, const _Float16* __restrict__ pA,
                const float* __restrict__ sc, const float* __restrict__ sh,
                const float* __restrict__ X, float* __restrict__ OUT) {
  constexpr int KC = 8;                        // 256 / 32
  __shared__ _Float16 ldsB[32 * LDSP];
  const int nt = blockIdx.x, n = blockIdx.y;
  const int p0 = nt * 32;
  const int t = threadIdx.x;
  const int lane = t & 31;
  const int mtBase = blockIdx.z * 16 + (t >> 5) * 4;   // global M-tile, 0..63
  const _Float16* H2n = H2 + (size_t)n * 256 * HW;
  const int k0  = (t >> 3) * 2;
  const int nc0 = (t & 7) * 4;
  v8f acc[4][2] = {};

  for (int kc = 0; kc < KC; ++kc) {
    const _Float16* row0 = H2n + (size_t)(kc * 32 + k0) * HW;
    const _Float16* row1 = row0 + HW;
    _Float16 a0[4], a1[4];
    if (p0 + 32 <= HW) {                       // 8B-aligned v4h fast path
      v4h u0 = *(const v4h*)(row0 + p0 + nc0);
      v4h u1 = *(const v4h*)(row1 + p0 + nc0);
#pragma unroll
      for (int j = 0; j < 4; ++j) { a0[j] = u0[j]; a1[j] = u1[j]; }
    } else {
#pragma unroll
      for (int j = 0; j < 4; ++j) {
        int p = p0 + nc0 + j;
        int pc = p < HW ? p : HW - 1;
        _Float16 x0 = row0[pc], x1 = row1[pc];
        a0[j] = p < HW ? x0 : (_Float16)0.0f;
        a1[j] = p < HW ? x1 : (_Float16)0.0f;
      }
    }
#pragma unroll
    for (int j = 0; j < 4; ++j) {
      v2h pr; pr[0] = a0[j]; pr[1] = a1[j];
      *(v2h*)&ldsB[(nc0 + j) * LDSP + k0] = pr;
    }
    __syncthreads();

    v16h bf0 = *(const v16h*)&ldsB[(lane & 15) * LDSP + ((lane >> 4) << 4)];
    v16h bf1 = *(const v16h*)&ldsB[(16 + (lane & 15)) * LDSP + ((lane >> 4) << 4)];
    const _Float16* aB = pA + ((size_t)mtBase * KC + kc) * 512 + lane * 16;
#pragma unroll
    for (int i = 0; i < 4; ++i) {
      v16h af = *(const v16h*)(aB + (size_t)i * KC * 512);
      acc[i][0] = wmma_f16(af, bf0, acc[i][0]);
      acc[i][1] = wmma_f16(af, bf1, acc[i][1]);
    }
    __syncthreads();
  }

  const int ncl = lane & 15;
  const int mh  = (lane >> 4) * 8;
  const bool ok0 = p0 + ncl < HW, ok1 = p0 + 16 + ncl < HW;
#pragma unroll
  for (int i = 0; i < 4; ++i)
#pragma unroll
    for (int r = 0; r < 8; ++r) {
      int cc = (mtBase + i) * 16 + mh + r;     // 0..1023
      float scv = sc[cc], shv = sh[cc];
      size_t o = ((size_t)n * 1024 + cc) * HW + p0 + ncl;
      if (ok0) {
        float y = acc[i][0][r] + 1.0f;
        OUT[o] = y * y * scv + shv + X[o];              // fused residual
      }
      if (ok1) {
        float y = acc[i][1][r] + 1.0f;
        OUT[o + 16] = y * y * scv + shv + X[o + 16];
      }
    }
}

// ============================================================================
extern "C" void kernel_launch(void* const* d_in, const int* in_sizes, int n_in,
                              void* d_out, int out_size, void* d_ws, size_t ws_size,
                              hipStream_t stream) {
  (void)in_sizes; (void)n_in; (void)out_size; (void)ws_size;
  const float* x  = (const float*)d_in[0];
  const float* w1 = (const float*)d_in[1];
  const float* w2 = (const float*)d_in[2];
  const float* w3 = (const float*)d_in[3];
  const float* g1 = (const float*)d_in[4];  const float* b1 = (const float*)d_in[5];
  const float* m1 = (const float*)d_in[6];  const float* v1 = (const float*)d_in[7];
  const float* g2 = (const float*)d_in[8];  const float* b2 = (const float*)d_in[9];
  const float* m2 = (const float*)d_in[10]; const float* v2 = (const float*)d_in[11];
  const float* g3 = (const float*)d_in[12]; const float* b3 = (const float*)d_in[13];
  const float* m3 = (const float*)d_in[14]; const float* v3 = (const float*)d_in[15];

  // workspace carve-out (256B aligned slabs), total ~27.9 MB
  char* ws = (char*)d_ws;
  size_t off = 0;
  auto carve = [&](size_t bytes) -> void* {
    void* p = ws + off;
    off = (off + bytes + 255) & ~(size_t)255;
    return p;
  };
  _Float16* pA1 = (_Float16*)carve((size_t)256 * 1024 * 2);
  _Float16* pA2 = (_Float16*)carve((size_t)256 * 2304 * 2);
  _Float16* pA3 = (_Float16*)carve((size_t)1024 * 256 * 2);
  float* sc1 = (float*)carve(256 * 4);   float* sh1 = (float*)carve(256 * 4);
  float* sc2 = (float*)carve(256 * 4);   float* sh2 = (float*)carve(256 * 4);
  float* sc3 = (float*)carve(1024 * 4);  float* sh3 = (float*)carve(1024 * 4);
  _Float16* h1 = (_Float16*)carve((size_t)128 * 256 * HW * 2);
  _Float16* h2 = (_Float16*)carve((size_t)128 * 256 * HW * 2);

  // prep: weight pre-pack into WMMA fragment order + BN folds (tiny)
  pack_w<<<(256 * 1024 + 255) / 256, 256, 0, stream>>>(w1, pA1, 32, 1024, 256 * 1024);
  pack_w<<<(256 * 2304 + 255) / 256, 256, 0, stream>>>(w2, pA2, 72, 2304, 256 * 2304);
  pack_w<<<(1024 * 256 + 255) / 256, 256, 0, stream>>>(w3, pA3, 8, 256, 1024 * 256);
  bn_fold<<<1, 256, 0, stream>>>(g1, b1, m1, v1, sc1, sh1, 256);
  bn_fold<<<1, 256, 0, stream>>>(g2, b2, m2, v2, sc2, sh2, 256);
  bn_fold<<<4, 256, 0, stream>>>(g3, b3, m3, v3, sc3, sh3, 1024);

  dim3 g12(NT32, 128, 1);
  kerv_gemm1<<<g12, 128, 0, stream>>>(x, pA1, sc1, sh1, h1);
  kerv_gemm2<<<g12, 128, 0, stream>>>(h1, pA2, sc2, sh2, h2);
  dim3 gg3(NT32, 128, 4);
  kerv_gemm3<<<gg3, 128, 0, stream>>>(h2, pA3, sc3, sh3, x, (float*)d_out);
}
